// MessagePassing_43782896615602
// MI455X (gfx1250) — compile-verified, matrix operand order checked by reading, not run on hardware
//
#include <hip/hip_runtime.h>

// ---------------------------------------------------------------------------
// Types for CDNA5 WMMA (wave32): v_wmma_f32_16x16x32_bf16
// ---------------------------------------------------------------------------
typedef __attribute__((ext_vector_type(16))) __bf16 v16bf;
typedef __attribute__((ext_vector_type(8)))  float  v8f;

union FragU { uint4 u[2]; v16bf v; };

struct MlpP {
  const __bf16* W1; const __bf16* W2; const __bf16* W3;
  const float* b1; const float* b2; const float* b3;
  const float* g;  const float* bt;
};

// A fragment (16x32 bf16, MxK) from LDS, row-major activation buffer.
// Lanes 0-15 hold row m, K = kb..kb+7 and kb+16..kb+23; lanes 16-31 hold
// K = kb+8..kb+15 and kb+24..kb+31 (two bf16 per VGPR).
__device__ __forceinline__ v16bf load_frag_a(const __bf16* lds, int rowBase,
                                             int stride, int kbase, int lane) {
  int m = lane & 15, gsel = lane >> 4;
  const __bf16* p = lds + (rowBase + m) * stride + kbase + gsel * 8;
  FragU f;
  f.u[0] = *(const uint4*)(p);
  f.u[1] = *(const uint4*)(p + 16);
  return f.v;
}

// B fragment (32x16 bf16, KxN) from the fragment-packed weight buffer:
// 32 lanes x 16 bf16 contiguous per fragment.
__device__ __forceinline__ v16bf load_frag_b(const __bf16* wbase, int frag, int lane) {
  const __bf16* p = wbase + ((long)frag * 32 + lane) * 16;
  FragU f;
  f.u[0] = *(const uint4*)(p);
  f.u[1] = *(const uint4*)(p + 8);
  return f.v;
}

__device__ __forceinline__ v8f wmma_bf16(v16bf a, v16bf b, v8f c) {
  return __builtin_amdgcn_wmma_f32_16x16x32_bf16(false, a, false, b,
                                                 (short)0, c, false, false);
}

// One MLP layer for a wave's 32-row slab (two 16-row A fragments share each
// B fragment). All 8 B fragments of a k-step are loaded as one clause so the
// compiler emits a single wait per 16 WMMAs. Bias+ReLU applied in place.
__device__ __forceinline__ void gemm_layer(const __bf16* xlds, int xstride,
                                           int ktiles, const __bf16* wfrag,
                                           const float* bias, bool relu,
                                           v8f c0[8], v8f c1[8],
                                           int lane, int rowBase) {
#pragma unroll
  for (int t = 0; t < 8; ++t)
#pragma unroll
    for (int i = 0; i < 8; ++i) { c0[t][i] = 0.0f; c1[t][i] = 0.0f; }

  for (int kt = 0; kt < ktiles; ++kt) {
    v16bf b[8];
#pragma unroll
    for (int t = 0; t < 8; ++t) b[t] = load_frag_b(wfrag, kt * 8 + t, lane);
    v16bf a0 = load_frag_a(xlds, rowBase,      xstride, kt * 32, lane);
    v16bf a1 = load_frag_a(xlds, rowBase + 16, xstride, kt * 32, lane);
#pragma unroll
    for (int t = 0; t < 8; ++t) {
      c0[t] = wmma_bf16(a0, b[t], c0[t]);
      c1[t] = wmma_bf16(a1, b[t], c1[t]);
    }
  }
  int n0 = lane & 15;
#pragma unroll
  for (int t = 0; t < 8; ++t) {
    float bv = bias[t * 16 + n0];
#pragma unroll
    for (int r = 0; r < 8; ++r) {
      float x0 = c0[t][r] + bv;
      float x1 = c1[t][r] + bv;
      c0[t][r] = relu ? fmaxf(x0, 0.0f) : x0;
      c1[t][r] = relu ? fmaxf(x1, 0.0f) : x1;
    }
  }
}

// D tiles (f32) -> bf16 row-major LDS (stride 128) for next layer's A frags.
// C/D layout: lane l holds col n0=l&15, rows (l>>4)*8 + r.
__device__ __forceinline__ void store_hidden(__bf16* hlds, int rowBase, int lane,
                                             const v8f c[8]) {
  int n0 = lane & 15, rofs = (lane >> 4) * 8;
#pragma unroll
  for (int t = 0; t < 8; ++t)
#pragma unroll
    for (int r = 0; r < 8; ++r)
      hlds[(rowBase + rofs + r) * 128 + t * 16 + n0] = (__bf16)c[t][r];
}

// LayerNorm over the 128-wide row, store to out, optional atomic segment-sum.
__device__ __forceinline__ void ln_store(const v8f c[8], const float* g,
                                         const float* bt, float* outp,
                                         float* effect, const int* recv,
                                         long rowGlobal, int lane, long Mlimit) {
  int n0 = lane & 15, rofs = (lane >> 4) * 8;
  float gv[8], bv[8];
#pragma unroll
  for (int t = 0; t < 8; ++t) { gv[t] = g[t * 16 + n0]; bv[t] = bt[t * 16 + n0]; }

#pragma unroll
  for (int r = 0; r < 8; ++r) {
    float s = 0.0f, ss = 0.0f, vals[8];
#pragma unroll
    for (int t = 0; t < 8; ++t) { float x = c[t][r]; vals[t] = x; s += x; ss += x * x; }
    // reduce across the 16 lanes holding this row (halves are independent rows)
#pragma unroll
    for (int m = 1; m < 16; m <<= 1) {
      s  += __shfl_xor(s,  m, 16);
      ss += __shfl_xor(ss, m, 16);
    }
    float mean = s * (1.0f / 128.0f);
    float var  = ss * (1.0f / 128.0f) - mean * mean;
    float rstd = __frsqrt_rn(var + 1e-5f);

    long erow = rowGlobal + rofs + r;
    bool inb = erow < Mlimit;
    int  rv  = (effect && inb) ? recv[erow] : 0;
#pragma unroll
    for (int t = 0; t < 8; ++t) {
      float y = (vals[t] - mean) * rstd * gv[t] + bv[t];
      int col = t * 16 + n0;
      if (inb) {
        outp[erow * 128 + col] = y;
        if (effect) atomicAdd(&effect[(long)rv * 128 + col], y);
      }
    }
  }
}

// Full 3-layer MLP + LN for a 64-row tile: 2 waves x 32 wave-private rows.
__device__ __forceinline__ void run_mlp(const __bf16* xlds, int xstride, int k1tiles,
                                        __bf16* hlds, const MlpP P, float* outp,
                                        float* effect, const int* recv,
                                        long blockRowBase, long Mlimit) {
  int lane = threadIdx.x & 31;
  int rowBase = (threadIdx.x >> 5) * 32;
  v8f c0[8], c1[8];
  gemm_layer(xlds, xstride, k1tiles, P.W1, P.b1, true, c0, c1, lane, rowBase);
  store_hidden(hlds, rowBase,      lane, c0);
  store_hidden(hlds, rowBase + 16, lane, c1);
  gemm_layer(hlds, 128, 4, P.W2, P.b2, true, c0, c1, lane, rowBase);
  store_hidden(hlds, rowBase,      lane, c0);
  store_hidden(hlds, rowBase + 16, lane, c1);
  gemm_layer(hlds, 128, 4, P.W3, P.b3, false, c0, c1, lane, rowBase);
  ln_store(c0, P.g, P.bt, outp, effect, recv, blockRowBase + rowBase,      lane, Mlimit);
  ln_store(c1, P.g, P.bt, outp, effect, recv, blockRowBase + rowBase + 16, lane, Mlimit);
}

// ---------------------------------------------------------------------------
// Kernels (64 threads = 2 waves per block; 64 rows per block)
// ---------------------------------------------------------------------------

// Edge kernel: shared gather of [node[s], node[r]] for normal + tangential MLPs.
__global__ __launch_bounds__(64) void edge_kernel(
    const float* __restrict__ nodef, const float* __restrict__ eN,
    const float* __restrict__ eT, const int* __restrict__ snd,
    const int* __restrict__ rcv, MlpP Pn, MlpP Pt, float* outN, float* outT,
    float* effect, int E) {
  __shared__ __bf16 xbuf[64 * 384];
  __shared__ __bf16 hbuf[64 * 128];
  long base = (long)blockIdx.x * 64;

  for (int i = threadIdx.x; i < 64 * 384; i += 64) {
    int r = i / 384, cidx = i - r * 384;
    long e = base + r; if (e >= E) e = E - 1;
    float v;
    if (cidx < 128)      v = nodef[(long)snd[e] * 128 + cidx];
    else if (cidx < 256) v = nodef[(long)rcv[e] * 128 + (cidx - 128)];
    else                 v = eN[e * 128 + (cidx - 256)];
    xbuf[r * 384 + cidx] = (__bf16)v;
  }
  __syncthreads();
  run_mlp(xbuf, 384, 12, hbuf, Pn, outN, effect, rcv, base, E);
  __syncthreads();
  // restage only the edge-feature columns for the tangential MLP
  for (int i = threadIdx.x; i < 64 * 128; i += 64) {
    int r = i >> 7, cidx = i & 127;
    long e = base + r; if (e >= E) e = E - 1;
    xbuf[r * 384 + 256 + cidx] = (__bf16)eT[e * 128 + cidx];
  }
  __syncthreads();
  run_mlp(xbuf, 384, 12, hbuf, Pt, outT, effect, rcv, base, E);
}

// Damping (self-edge) kernel: single MLP, K=384.
__global__ __launch_bounds__(64) void damp_kernel(
    const float* __restrict__ nodef, const float* __restrict__ eD,
    const int* __restrict__ snd, const int* __restrict__ rcv, MlpP Pd,
    float* outD, float* effect, int S) {
  __shared__ __bf16 xbuf[64 * 384];
  __shared__ __bf16 hbuf[64 * 128];
  long base = (long)blockIdx.x * 64;
  for (int i = threadIdx.x; i < 64 * 384; i += 64) {
    int r = i / 384, cidx = i - r * 384;
    long e = base + r; if (e >= S) e = S - 1;
    float v;
    if (cidx < 128)      v = nodef[(long)snd[e] * 128 + cidx];
    else if (cidx < 256) v = nodef[(long)rcv[e] * 128 + (cidx - 128)];
    else                 v = eD[e * 128 + (cidx - 256)];
    xbuf[r * 384 + cidx] = (__bf16)v;
  }
  __syncthreads();
  run_mlp(xbuf, 384, 12, hbuf, Pd, outD, effect, rcv, base, S);
}

// Node kernel: input [effect, node], K=256. effect aliases the out_node region;
// each block reads only its own rows before overwriting them.
__global__ __launch_bounds__(64) void node_kernel(
    const float* __restrict__ nodef, const float* __restrict__ effect, MlpP Pm,
    float* outNode, int N) {
  __shared__ __bf16 xbuf[64 * 256];
  __shared__ __bf16 hbuf[64 * 128];
  long base = (long)blockIdx.x * 64;
  for (int i = threadIdx.x; i < 64 * 256; i += 64) {
    int r = i >> 8, cidx = i & 255;
    long nrow = base + r; if (nrow >= N) nrow = N - 1;
    float v = (cidx < 128) ? effect[nrow * 128 + cidx]
                           : nodef[nrow * 128 + (cidx - 128)];
    xbuf[r * 256 + cidx] = (__bf16)v;
  }
  __syncthreads();
  run_mlp(xbuf, 256, 8, hbuf, Pm, outNode, nullptr, nullptr, base, N);
}

// Convert one f32 weight matrix [K x 128] into fragment-packed bf16.
// blockIdx = fragment (kt*8 + ntile); lane layout mirrors the WMMA B layout.
__global__ __launch_bounds__(32) void prep_w(const float* __restrict__ W,
                                             __bf16* __restrict__ dst, int ktiles) {
  int frag = blockIdx.x, lane = threadIdx.x;
  int kt = frag >> 3, t = frag & 7;
  int n = t * 16 + (lane & 15);
  int gsel = lane >> 4;
  __bf16* out = dst + ((long)frag * 32 + lane) * 16;
#pragma unroll
  for (int v = 0; v < 8; ++v) {
    int k = kt * 32 + ((v < 4) ? v * 2 : 16 + (v - 4) * 2) + gsel * 8;
    out[v * 2 + 0] = (__bf16)W[(long)k * 128 + n];
    out[v * 2 + 1] = (__bf16)W[(long)(k + 1) * 128 + n];
  }
}

__global__ void zero_kernel(float* p, long n) {
  long i = (long)blockIdx.x * blockDim.x + threadIdx.x;
  long stride = (long)gridDim.x * blockDim.x;
  for (; i < n; i += stride) p[i] = 0.0f;
}

// ---------------------------------------------------------------------------
// Launch
// ---------------------------------------------------------------------------
extern "C" void kernel_launch(void* const* d_in, const int* in_sizes, int n_in,
                              void* d_out, int out_size, void* d_ws, size_t ws_size,
                              hipStream_t stream) {
  const float* nodef = (const float*)d_in[0];
  const float* eN    = (const float*)d_in[1];
  const float* eT    = (const float*)d_in[2];
  const float* eD    = (const float*)d_in[3];
  const int* snd     = (const int*)d_in[4];
  const int* rcv     = (const int*)d_in[5];
  const int* se_s    = (const int*)d_in[6];
  const int* se_r    = (const int*)d_in[7];

  int N = in_sizes[0] / 128;
  int E = in_sizes[4];
  int S = in_sizes[6];

  float* out       = (float*)d_out;
  float* out_node  = out;                         // also the effect accumulator
  float* out_norm  = out + (long)N * 128;
  float* out_tang  = out_norm + (long)E * 128;
  float* out_damp  = out_tang + (long)E * 128;
  float* effect    = out_node;

  // bf16 fragment-packed weights in workspace (~623 KB total)
  __bf16* ws = (__bf16*)d_ws;
  __bf16* neW1 = ws;               // 12*8 frags * 512
  __bf16* neW2 = neW1 + 49152;     // 4*8 * 512
  __bf16* neW3 = neW2 + 16384;
  __bf16* teW1 = neW3 + 16384;
  __bf16* teW2 = teW1 + 49152;
  __bf16* teW3 = teW2 + 16384;
  __bf16* deW1 = teW3 + 16384;
  __bf16* deW2 = deW1 + 49152;
  __bf16* deW3 = deW2 + 16384;
  __bf16* nmW1 = deW3 + 16384;     // 8*8 * 512
  __bf16* nmW2 = nmW1 + 32768;
  __bf16* nmW3 = nmW2 + 16384;

  zero_kernel<<<2048, 256, 0, stream>>>(effect, (long)N * 128);

  prep_w<<<96, 32, 0, stream>>>((const float*)d_in[8],  neW1, 12);
  prep_w<<<32, 32, 0, stream>>>((const float*)d_in[10], neW2, 4);
  prep_w<<<32, 32, 0, stream>>>((const float*)d_in[12], neW3, 4);
  prep_w<<<96, 32, 0, stream>>>((const float*)d_in[16], teW1, 12);
  prep_w<<<32, 32, 0, stream>>>((const float*)d_in[18], teW2, 4);
  prep_w<<<32, 32, 0, stream>>>((const float*)d_in[20], teW3, 4);
  prep_w<<<96, 32, 0, stream>>>((const float*)d_in[24], deW1, 12);
  prep_w<<<32, 32, 0, stream>>>((const float*)d_in[26], deW2, 4);
  prep_w<<<32, 32, 0, stream>>>((const float*)d_in[28], deW3, 4);
  prep_w<<<64, 32, 0, stream>>>((const float*)d_in[32], nmW1, 8);
  prep_w<<<32, 32, 0, stream>>>((const float*)d_in[34], nmW2, 4);
  prep_w<<<32, 32, 0, stream>>>((const float*)d_in[36], nmW3, 4);

  MlpP Pn = { neW1, neW2, neW3, (const float*)d_in[9],  (const float*)d_in[11],
              (const float*)d_in[13], (const float*)d_in[14], (const float*)d_in[15] };
  MlpP Pt = { teW1, teW2, teW3, (const float*)d_in[17], (const float*)d_in[19],
              (const float*)d_in[21], (const float*)d_in[22], (const float*)d_in[23] };
  MlpP Pd = { deW1, deW2, deW3, (const float*)d_in[25], (const float*)d_in[27],
              (const float*)d_in[29], (const float*)d_in[30], (const float*)d_in[31] };
  MlpP Pm = { nmW1, nmW2, nmW3, (const float*)d_in[33], (const float*)d_in[35],
              (const float*)d_in[37], (const float*)d_in[38], (const float*)d_in[39] };

  edge_kernel<<<(E + 63) / 64, 64, 0, stream>>>(nodef, eN, eT, snd, rcv,
                                                Pn, Pt, out_norm, out_tang,
                                                effect, E);
  damp_kernel<<<(S + 63) / 64, 64, 0, stream>>>(nodef, eD, se_s, se_r, Pd,
                                                out_damp, effect, S);
  node_kernel<<<(N + 63) / 64, 64, 0, stream>>>(nodef, effect, Pm, out_node, N);
}
